// GraphConvolution_14216341750332
// MI455X (gfx1250) — matrix operand order repaired and checked by reading, not compiled, not run on hardware
//
#include <hip/hip_runtime.h>

typedef __attribute__((ext_vector_type(16))) _Float16 v16h;
typedef __attribute__((ext_vector_type(8)))  _Float16 v8h;
typedef __attribute__((ext_vector_type(8)))  float    v8f;
typedef __attribute__((ext_vector_type(4)))  float    v4f;

#define B_ 8
#define N_ 2048
#define F_ 64
#define O_ 64

// ---------------------------------------------------------------------------
// Phase 1: XW = X @ W  (tiny: 134 MFLOP). Store transposed [B][O][N] as an
// exact f16 hi/lo split so phase 2 can load WMMA B-operands contiguously.
// ---------------------------------------------------------------------------
__global__ __launch_bounds__(256) void gcn_xw_kernel(
    const float* __restrict__ X, const float* __restrict__ W,
    _Float16* __restrict__ xwt_hi, _Float16* __restrict__ xwt_lo)
{
  int t = blockIdx.x * 256 + threadIdx.x;       // t = ((b*O + o)*N + m)
  int m = t & (N_ - 1);
  int o = (t >> 11) & (O_ - 1);
  int b = t >> 17;

  const float* xr = X + ((size_t)b * N_ + m) * F_;
  float s = 0.f;
#pragma unroll
  for (int f = 0; f < F_; f += 4) {
    v4f xv = *(const v4f*)(xr + f);
    s += xv.x * W[(f + 0) * O_ + o];
    s += xv.y * W[(f + 1) * O_ + o];
    s += xv.z * W[(f + 2) * O_ + o];
    s += xv.w * W[(f + 3) * O_ + o];
  }
  _Float16 hi = (_Float16)s;
  _Float16 lo = (_Float16)(s - (float)hi);      // exact residual (fits f16)
  size_t idx = ((size_t)b * O_ + o) * N_ + m;   // coalesced over m
  xwt_hi[idx] = hi;
  xwt_lo[idx] = lo;
}

// ---------------------------------------------------------------------------
// Phase 2: out[b,n,:] = (sum_m A[b,n,m]*XW[b,m,:] + XW[b,n,:]) / (deg+1)
// One wave per 16x64 output tile. A streamed once from HBM (memory-bound),
// converted to split-f16 on the fly; 3x v_wmma_f32_16x16x32_f16 per tile pair
// recovers ~fp32 accuracy. Row degree accumulated in f32 from the same loads.
// ---------------------------------------------------------------------------
__global__ __launch_bounds__(256) void gcn_main_kernel(
    const float* __restrict__ A,
    const _Float16* __restrict__ xwt_hi,
    const _Float16* __restrict__ xwt_lo,
    float* __restrict__ out)
{
  const int wid  = (blockIdx.x * 256 + threadIdx.x) >> 5;  // 0..1023
  const int lane = threadIdx.x & 31;
  const int half = lane >> 4;        // 0: lanes 0-15, 1: lanes 16-31
  const int r    = lane & 15;
  const int b    = wid >> 7;
  const int m0   = (wid & 127) << 4; // 16-row tile base

  const float* arow = A + ((size_t)b * N_ + (m0 + r)) * N_;
  const _Float16* bh_base = xwt_hi + (size_t)b * O_ * N_;
  const _Float16* bl_base = xwt_lo + (size_t)b * O_ * N_;

  v8f c[4];
#pragma unroll
  for (int ot = 0; ot < 4; ++ot) c[ot] = (v8f){};
  float degp = 0.f;

  for (int k0 = 0; k0 < N_; k0 += 32) {
    // ---- A tile 16x32, contiguous k order (same as B):
    // lane group 0 holds K = 0..15, group 1 holds K = 16..31
    v4f a0 = *(const v4f*)(arow + k0 + half * 16);
    v4f a1 = *(const v4f*)(arow + k0 + half * 16 + 4);
    v4f a2 = *(const v4f*)(arow + k0 + half * 16 + 8);
    v4f a3 = *(const v4f*)(arow + k0 + half * 16 + 12);
    __builtin_prefetch(arow + k0 + 256, 0, 1);   // global_prefetch_b8

    v16h ahi, alo;
#pragma unroll
    for (int i = 0; i < 4; ++i) {
      float f0 = a0[i], f1 = a1[i], f2 = a2[i], f3 = a3[i];
      degp += f0 + f1 + f2 + f3;                 // row-degree partial (f32)
      _Float16 h0 = (_Float16)f0, h1 = (_Float16)f1;
      _Float16 h2 = (_Float16)f2, h3 = (_Float16)f3;
      ahi[i]      = h0;  alo[i]      = (_Float16)(f0 - (float)h0);
      ahi[4 + i]  = h1;  alo[4 + i]  = (_Float16)(f1 - (float)h1);
      ahi[8 + i]  = h2;  alo[8 + i]  = (_Float16)(f2 - (float)h2);
      ahi[12 + i] = h3;  alo[12 + i] = (_Float16)(f3 - (float)h3);
    }

    // ---- 4 o-tiles of 16 columns; B operand: lanes 0-15 K=0..15,
    // lanes 16-31 K=16..31 -> contiguous 16 halfs from transposed XW.
#pragma unroll
    for (int ot = 0; ot < 4; ++ot) {
      size_t boff = (size_t)(ot * 16 + r) * N_ + k0 + half * 16;
      v8h bh0 = *(const v8h*)(bh_base + boff);
      v8h bh1 = *(const v8h*)(bh_base + boff + 8);
      v8h bl0 = *(const v8h*)(bl_base + boff);
      v8h bl1 = *(const v8h*)(bl_base + boff + 8);
      v16h bhi = __builtin_shufflevector(bh0, bh1, 0,1,2,3,4,5,6,7,
                                                   8,9,10,11,12,13,14,15);
      v16h blo = __builtin_shufflevector(bl0, bl1, 0,1,2,3,4,5,6,7,
                                                   8,9,10,11,12,13,14,15);
      // split-f16 product: hi*hi + hi*lo + lo*hi  (lo*lo ~ 2^-22, dropped)
      c[ot] = __builtin_amdgcn_wmma_f32_16x16x32_f16(
          false, ahi, false, bhi, (short)0, c[ot], false, false);
      c[ot] = __builtin_amdgcn_wmma_f32_16x16x32_f16(
          false, ahi, false, blo, (short)0, c[ot], false, false);
      c[ot] = __builtin_amdgcn_wmma_f32_16x16x32_f16(
          false, alo, false, bhi, (short)0, c[ot], false, false);
    }
  }

  // ---- degree for row r: lane r covered K = {0..15,32..47,...},
  // lane r+16 covered the complementary halves.
  float deg = degp + __shfl_xor(degp, 16, 32) + 1.0f;  // +1: self loop

  // ---- epilogue: C layout VGPR v -> M = v + half*8, N-col = r
#pragma unroll
  for (int v = 0; v < 8; ++v) {
    int M = v + half * 8;
    float dv  = __shfl(deg, M, 32);   // lane M holds deg for row M
    float rdv = 1.0f / dv;
    size_t orow = ((size_t)b * N_ + m0 + M) * O_;
#pragma unroll
    for (int ot = 0; ot < 4; ++ot) {
      int o = ot * 16 + r;
      size_t didx = ((size_t)b * O_ + o) * N_ + (m0 + M);
      float diag = (float)xwt_hi[didx] + (float)xwt_lo[didx]; // self-loop XW
      out[orow + o] = (c[ot][v] + diag) * rdv;
    }
  }
}

// ---------------------------------------------------------------------------
extern "C" void kernel_launch(void* const* d_in, const int* in_sizes, int n_in,
                              void* d_out, int out_size, void* d_ws, size_t ws_size,
                              hipStream_t stream) {
  const float* A = (const float*)d_in[0];   // [8,2048,2048]
  const float* X = (const float*)d_in[1];   // [8,2048,64]
  const float* W = (const float*)d_in[2];   // [64,64]
  float* out = (float*)d_out;               // [8,2048,64]

  _Float16* xwt_hi = (_Float16*)d_ws;                          // 2 MB
  _Float16* xwt_lo = xwt_hi + (size_t)B_ * O_ * N_;            // 2 MB

  gcn_xw_kernel<<<(B_ * O_ * N_) / 256, 256, 0, stream>>>(X, W, xwt_hi, xwt_lo);

  // 1024 waves (one per 16-row tile per batch) = 32768 threads = 128 blocks
  gcn_main_kernel<<<(B_ * (N_ / 16) * 32) / 256, 256, 0, stream>>>(
      A, xwt_hi, xwt_lo, out);
}
